// MemoryReader_80049600463577
// MI455X (gfx1250) — matrix operand order repaired and checked by reading, not compiled, not run on hardware
//
#include <hip/hip_runtime.h>
#include <hip/hip_bf16.h>
#include <math.h>

// ---------------- CDNA5 vector types ----------------
typedef __attribute__((ext_vector_type(16))) __bf16          v16bf;
typedef __attribute__((ext_vector_type(8)))  float           v8f;
typedef __attribute__((ext_vector_type(2)))  float           v2f;
typedef __attribute__((ext_vector_type(4)))  unsigned short  us4;
typedef __attribute__((ext_vector_type(8)))  unsigned short  us8;
typedef __attribute__((ext_vector_type(16))) unsigned short  us16;
typedef __attribute__((ext_vector_type(4)))  unsigned int    u32x4;
typedef __attribute__((ext_vector_type(4)))  int             i32x4;
typedef __attribute__((ext_vector_type(8)))  int             i32x8;

// ---------------- problem constants ----------------
#define BATCH 4
#define CK    64
#define CV    128
#define MM    9216      // N*H*W memory locations
#define QQ    2304      // H*W query locations
#define OC    64
#define IC    256
#define MT    32        // memory tile per iteration (WMMA K for PV)
#define QB    64        // queries per block (4 q-tiles of 16)
#define NSPLIT 4        // M-splits per block (waves along M)
#define MPART (MM / NSPLIT)

// TDM available? (device pass only; arity differs between clang 22 / 23)
#if defined(__AMDGCN__) && __has_builtin(__builtin_amdgcn_tensor_load_to_lds) && \
    __has_builtin(__builtin_amdgcn_s_wait_tensorcnt)
#define USE_TDM 1
#else
#define USE_TDM 0
#endif

__device__ __forceinline__ unsigned short f2bf(float f) {
  unsigned int u = __builtin_bit_cast(unsigned int, f);
  u += 0x7FFFu + ((u >> 16) & 1u);           // round-to-nearest-even
  return (unsigned short)(u >> 16);
}

__device__ __forceinline__ v16bf mk16(us8 lo, us8 hi8) {
  us16 t = __builtin_shufflevector(lo, hi8, 0,1,2,3,4,5,6,7,8,9,10,11,12,13,14,15);
  return __builtin_bit_cast(v16bf, t);
}

__global__ __launch_bounds__(512, 1) void mr_fused_kernel(
    const float* __restrict__ KM, const float* __restrict__ VM,
    const float* __restrict__ KQ, const float* __restrict__ VQ,
    const float* __restrict__ Wc, const float* __restrict__ bnG,
    const float* __restrict__ bnB, const float* __restrict__ bnM,
    const float* __restrict__ bnV, float* __restrict__ out)
{
  // per-M-split bf16 staging tiles
  __shared__ __align__(16) unsigned short sK[NSPLIT][MT][CK];   // Km^T: [m][c]
  __shared__ __align__(16) unsigned short sV[NSPLIT][CV][MT];   // Vm:   [c][m]
  // fp32 concat(mem, V_Q) for the fused conv epilogue
  __shared__ __align__(16) float sE[IC][QB];                    // 64 KB
  // cross-split softmax stats: [q-tile][split][q]
  __shared__ float sMax[4][NSPLIT][16];
  __shared__ float sSum[4][NSPLIT][16];

  const int tid  = threadIdx.x;
  const int wave = tid >> 5;            // 0..15
  const int lane = tid & 31;
  const int ln   = lane & 15;
  const int hsel = (lane >> 4) & 1;     // 0: lanes 0-15, 1: lanes 16-31
  const int qt   = wave & 3;            // q-tile within block
  const int sp   = wave >> 2;           // M-split 0..3
  const int lt   = tid & 127;           // thread within split group

  const int b     = blockIdx.x / (QQ / QB);
  const int q_blk = (blockIdx.x % (QQ / QB)) * QB;
  const int q0    = q_blk + qt * 16;
  const int qcol  = q0 + ln;

#if USE_TDM
  // ---- kick off TDM: V_Q tile [128 c x 64 q] -> sE rows [128,256); overlaps
  // the whole attention phase.  D# per CDNA5 ISA.
  if (wave == 0) {
    unsigned lds_off = (unsigned)(size_t)(&sE[CV][0]);
    unsigned long long ga =
        (unsigned long long)(size_t)(VQ + (size_t)b * CV * QQ + q_blk);
    u32x4 g0;
    g0[0] = 1u;                                    // count=1 (valid user D#)
    g0[1] = lds_off;                               // lds_addr (bytes)
    g0[2] = (unsigned)(ga & 0xffffffffu);          // global_addr[31:0]
    g0[3] = (unsigned)((ga >> 32) & 0x01ffffffu)   // global_addr[56:32]
          | 0x80000000u;                           // type=2 ("image")
    i32x8 g1;
    g1[0] = 0x00020000;                            // data_size = 2 -> 4 bytes
    g1[1] = (QQ & 0xffff) << 16;                   // tensor_dim0 lo16
    g1[2] = ((CV & 0xffff) << 16) | ((QQ >> 16) & 0xffff);  // dim1 lo | dim0 hi
    g1[3] = (QB << 16) | ((CV >> 16) & 0xffff);    // tile_dim0=64 | dim1 hi
    g1[4] = CV;                                    // tile_dim1=128 (tile_dim2=0)
    g1[5] = QQ;                                    // tensor_dim0_stride lo32
    g1[6] = 0;
    g1[7] = 0;
    i32x4 gz; gz[0] = gz[1] = gz[2] = gz[3] = 0;
#if __clang_major__ >= 23
    i32x8 gz8; for (int i = 0; i < 8; ++i) gz8[i] = 0;
    __builtin_amdgcn_tensor_load_to_lds(g0, g1, gz, gz, gz8, 0);
#else
    __builtin_amdgcn_tensor_load_to_lds(g0, g1, gz, gz, 0);
#endif
  }
#endif

  // ---- Kq B-operands (B layout: lane<16 -> K=c 0..7,16..23, lane>=16 -> K=c 8..15,24..31)
  v16bf bq[2];
  for (int j = 0; j < 2; ++j) {
    us16 t;
    for (int h = 0; h < 16; ++h) {
      int c = j * 32 + (h < 8 ? hsel * 8 + h : 16 + hsel * 8 + (h - 8));
      t[h] = f2bf(KQ[((size_t)b * CK + c) * QQ + qcol]);
    }
    bq[j] = __builtin_bit_cast(v16bf, t);
  }

  // flash accumulators for this wave's (q-tile, M-split)
  v8f acc[8];
  for (int i = 0; i < 8; ++i)
    for (int r = 0; r < 8; ++r) acc[i][r] = 0.0f;
  float runmax = -3.0e38f, runsum = 0.0f;

  const int mbase = sp * MPART;
  for (int it = 0; it < MPART / MT; ++it) {
    const int im = mbase + it * MT;

    // stage Km tile transposed [m][c] bf16 (b128 global loads over m)
    for (int e = lt; e < CK * (MT / 4); e += 128) {
      int c = e >> 3, m4 = (e & 7) << 2;
      float4 v = *(const float4*)&KM[((size_t)b * CK + c) * MM + im + m4];
      sK[sp][m4 + 0][c] = f2bf(v.x);
      sK[sp][m4 + 1][c] = f2bf(v.y);
      sK[sp][m4 + 2][c] = f2bf(v.z);
      sK[sp][m4 + 3][c] = f2bf(v.w);
    }
    // stage Vm tile [c][m] bf16 (b128 loads, packed b64 LDS stores)
    for (int e = lt; e < CV * (MT / 4); e += 128) {
      int c = e >> 3, m4 = (e & 7) << 2;
      float4 v = *(const float4*)&VM[((size_t)b * CV + c) * MM + im + m4];
      us4 pk;
      pk[0] = f2bf(v.x); pk[1] = f2bf(v.y); pk[2] = f2bf(v.z); pk[3] = f2bf(v.w);
      *(us4*)&sV[sp][c][m4] = pk;
    }
    __syncthreads();

    // ---- scores S[t][m,q] = Km^T Kq over 64 channels (2 K-steps of 32) ----
    v8f S[2];
    for (int t = 0; t < 2; ++t) {
      v8f s = {0.f,0.f,0.f,0.f,0.f,0.f,0.f,0.f};
      const us8* rowp = (const us8*)(&sK[sp][t * 16 + ln][0]);
      for (int j = 0; j < 2; ++j) {
        v16bf a = mk16(rowp[j * 4 + hsel], rowp[j * 4 + 2 + hsel]);
        s = __builtin_amdgcn_wmma_f32_16x16x32_bf16(false, a, false, bq[j],
                                                    (short)0, s, false, false);
      }
      S[t] = s;
    }

    // ---- online softmax over m (per q column) ----
    float tmax = -3.0e38f;
    for (int r = 0; r < 8; ++r) {
      S[0][r] *= 0.125f;  // 1/sqrt(64)
      S[1][r] *= 0.125f;
      tmax = fmaxf(tmax, fmaxf(S[0][r], S[1][r]));
    }
    tmax = fmaxf(tmax, __shfl_xor(tmax, 16, 32));
    float nmax = fmaxf(runmax, tmax);
    float rf   = __expf(runmax - nmax);
    runmax = nmax;
    float ls = 0.0f;
    for (int r = 0; r < 8; ++r) {
      S[0][r] = __expf(S[0][r] - nmax);
      S[1][r] = __expf(S[1][r] - nmax);
      ls += S[0][r] + S[1][r];
    }
    ls += __shfl_xor(ls, 16, 32);
    runsum = runsum * rf + ls;
    for (int i = 0; i < 8; ++i)
      for (int r = 0; r < 8; ++r) acc[i][r] *= rf;

    // ---- repack P (D layout) into bf16 B-operand (K = m 0..31) ----
    us16 pb;
    for (int r = 0; r < 8; ++r) {
      float o0 = __shfl_xor(S[0][r], 16, 32);
      float o1 = __shfl_xor(S[1][r], 16, 32);
      pb[r]     = f2bf(hsel ? o1      : S[0][r]);
      pb[8 + r] = f2bf(hsel ? S[1][r] : o0);
    }
    v16bf bp = __builtin_bit_cast(v16bf, pb);

    // ---- PV: acc[ci] += Vm[16c x 32m] x P[32m x 16q] ----
    for (int ci = 0; ci < 8; ++ci) {
      const us8* rp = (const us8*)(&sV[sp][ci * 16 + ln][0]);
      v16bf a = mk16(rp[2 * hsel], rp[2 * hsel + 1]);
      acc[ci] = __builtin_amdgcn_wmma_f32_16x16x32_bf16(false, a, false, bp,
                                                        (short)0, acc[ci], false, false);
    }
    __syncthreads();
  }

  // ---- cross-split softmax merge (deterministic) ----
  if (!hsel) { sMax[qt][sp][ln] = runmax; sSum[qt][sp][ln] = runsum; }
  __syncthreads();
  float gm = fmaxf(fmaxf(sMax[qt][0][ln], sMax[qt][1][ln]),
                   fmaxf(sMax[qt][2][ln], sMax[qt][3][ln]));
  float tot = 0.0f;
  for (int s = 0; s < NSPLIT; ++s)
    tot += sSum[qt][s][ln] * __expf(sMax[qt][s][ln] - gm);
  const float wsc = __expf(runmax - gm) / tot;

  // barriered sequential accumulation into sE rows [0,128) — fixed order
  for (int s = 0; s < NSPLIT; ++s) {
    if (sp == s) {
      for (int ci = 0; ci < 8; ++ci)
        for (int r = 0; r < 8; ++r) {
          int c = ci * 16 + r + hsel * 8;
          float v = acc[ci][r] * wsc;
          if (s == 0) sE[c][qt * 16 + ln] = v;
          else        sE[c][qt * 16 + ln] += v;
        }
    }
    __syncthreads();
  }

#if USE_TDM
  if (wave == 0) __builtin_amdgcn_s_wait_tensorcnt(0);
#else
  for (int e = tid; e < CV * QB; e += 512) {
    int c = e >> 6, qi = e & 63;
    sE[CV + c][qi] = VQ[((size_t)b * CV + c) * QQ + q_blk + qi];
  }
#endif
  __syncthreads();

  // ---- fused conv1x1 (f32 WMMA 16x16x4) + BN + ReLU: one tile per wave ----
  const int ot   = wave & 3;          // output-channel tile
  const int qt2  = wave >> 2;         // query tile
  const int orow = ot * 16 + ln;      // A row = output channel
  const int qloc = qt2 * 16 + ln;
  v8f d = {0.f,0.f,0.f,0.f,0.f,0.f,0.f,0.f};
  for (int c = 0; c < IC; c += 4) {
    int ca = c + hsel * 2;            // f32 A/B: lanes 0-15 K={0,1}, 16-31 K={2,3}
    v2f a, bb2;
    a.x   = Wc[(size_t)orow * IC + ca];
    a.y   = Wc[(size_t)orow * IC + ca + 1];
    bb2.x = sE[ca][qloc];
    bb2.y = sE[ca + 1][qloc];
    d = __builtin_amdgcn_wmma_f32_16x16x4_f32(false, a, false, bb2,
                                              (short)0, d, false, false);
  }
  for (int r = 0; r < 8; ++r) {
    int o = ot * 16 + r + hsel * 8;
    float is = rsqrtf(bnV[o] + 1e-5f);
    float sc = bnG[o] * is;
    float sh = bnB[o] - bnG[o] * bnM[o] * is;
    float v  = d[r] * sc + sh;
    v = v > 0.0f ? v : 0.0f;
    out[((size_t)b * OC + o) * QQ + q_blk + qt2 * 16 + ln] = v;
  }
}

extern "C" void kernel_launch(void* const* d_in, const int* in_sizes, int n_in,
                              void* d_out, int out_size, void* d_ws, size_t ws_size,
                              hipStream_t stream) {
  (void)in_sizes; (void)n_in; (void)d_ws; (void)ws_size; (void)out_size;
  const float* KM = (const float*)d_in[0];
  const float* VM = (const float*)d_in[1];
  const float* KQ = (const float*)d_in[2];
  const float* VQ = (const float*)d_in[3];
  const float* Wc = (const float*)d_in[4];
  const float* g  = (const float*)d_in[5];
  const float* be = (const float*)d_in[6];
  const float* mu = (const float*)d_in[7];
  const float* va = (const float*)d_in[8];
  float* out = (float*)d_out;

  dim3 grid(BATCH * (QQ / QB));   // 144 blocks
  dim3 block(512);                // 16 waves: 4 q-tiles x 4 M-splits
  mr_fused_kernel<<<grid, block, 0, stream>>>(KM, VM, KQ, VQ, Wc, g, be, mu, va, out);
}
